// QcMarianAttention_42563125903657
// MI455X (gfx1250) — matrix operand order repaired and checked by reading, not compiled, not run on hardware
//
#include <hip/hip_runtime.h>

typedef __attribute__((ext_vector_type(16))) _Float16 v16h;
typedef __attribute__((ext_vector_type(8)))  _Float16 v8h;
typedef __attribute__((ext_vector_type(8)))  float    v8f;
typedef __attribute__((ext_vector_type(4)))  float    f4;
typedef __attribute__((ext_vector_type(4)))  unsigned int u32x4;
typedef __attribute__((ext_vector_type(8)))  int      i32x8;
typedef __attribute__((ext_vector_type(4)))  int      i32x4;

static constexpr int B_ = 4;
static constexpr int T_ = 1024;
static constexpr int E_ = 1024;
static constexpr int H_ = 16;
static constexpr int D_ = 64;

__device__ __forceinline__ v16h cat8(v8h lo, v8h hi) {
  v16h r;
#pragma unroll
  for (int i = 0; i < 8; ++i) { r[i] = lo[i]; r[i + 8] = hi[i]; }
  return r;
}

__device__ __forceinline__ v8f wmma_f16(v16h a, v16h b, v8f c) {
  // (neg_a, A, neg_b, B, c_mod, C, reuse_a, reuse_b)
  return __builtin_amdgcn_wmma_f32_16x16x32_f16(false, a, false, b, (short)0, c,
                                                false, false);
}

// LDS byte offset of a __shared__ generic pointer (aperture low 32 bits).
__device__ __forceinline__ unsigned lds_off(const void* p) {
  return (unsigned)(unsigned long long)p;
}

// ---------------------------------------------------------------------------
// Tensor Data Mover: 2D tile (16-bit elements) Global -> LDS.
// D# packing per CDNA5 ISA ch.8: group0 = {count/type/addr}, group1 =
// {data_size, tensor dims, tile dims, strides}.  Groups 2/3 zero (<=2D).
// ---------------------------------------------------------------------------
__device__ __forceinline__ void tdm_load_2d_f16(
    unsigned lds_byte, unsigned long long gaddr, unsigned tile_d0,
    unsigned tile_d1, unsigned tensor_d0, unsigned tensor_d1,
    unsigned long long stride0_elems) {
  u32x4 g0;
  g0[0] = 1u;                                   // count=1, user mode
  g0[1] = lds_byte;                             // lds_addr
  g0[2] = (unsigned)(gaddr & 0xffffffffull);    // global_addr[31:0]
  g0[3] = (unsigned)((gaddr >> 32) & 0x1ffffffull) | (2u << 30);  // addr hi|type=2
  i32x8 g1;
  g1[0] = (int)(1u << 16);                      // workgroup_mask=0, data_size=1 (2B)
  g1[1] = (int)((tensor_d0 & 0xffffu) << 16);   // tensor_dim0[15:0]
  g1[2] = (int)((tensor_d0 >> 16) | ((tensor_d1 & 0xffffu) << 16));
  g1[3] = (int)((tensor_d1 >> 16) | ((tile_d0 & 0xffffu) << 16));
  g1[4] = (int)(tile_d1 & 0xffffu);             // tile_dim1, tile_dim2=0
  g1[5] = (int)(unsigned)(stride0_elems & 0xffffffffull);
  g1[6] = (int)(unsigned)((stride0_elems >> 32) & 0xffffull);  // dim1_stride[15:0]=0
  g1[7] = 0;
  i32x4 z4 = {0, 0, 0, 0};
#if defined(__clang_major__) && (__clang_major__ >= 23)
  i32x8 z8 = {0, 0, 0, 0, 0, 0, 0, 0};
  __builtin_amdgcn_tensor_load_to_lds(g0, g1, z4, z4, z8, 0);
#else
  __builtin_amdgcn_tensor_load_to_lds(g0, g1, z4, z4, 0);
#endif
}

// ---------------------------------------------------------------------------
// Kernel 1: fused QKV projection.  out = X @ W^T + b  (Q additionally *0.125)
// Block: 256 thr = 8 waves (4x2), tile 128x128, wave tile 32x64 (2x4 WMMA).
// Q,K stored [B,H,T,D] f16; V stored transposed [B,H,D,T].
// ---------------------------------------------------------------------------
__global__ __launch_bounds__(256) void qkv_proj_kernel(
    const float* __restrict__ X,
    const float* __restrict__ Wq, const float* __restrict__ bq,
    const float* __restrict__ Wk, const float* __restrict__ bk,
    const float* __restrict__ Wv, const float* __restrict__ bv,
    _Float16* __restrict__ Qo, _Float16* __restrict__ Ko,
    _Float16* __restrict__ Vt) {
  __shared__ _Float16 As[128 * 32];
  __shared__ _Float16 Bs[128 * 32];

  const int which = blockIdx.z;
  const float* __restrict__ W  = (which == 0) ? Wq : (which == 1 ? Wk : Wv);
  const float* __restrict__ bb = (which == 0) ? bq : (which == 1 ? bk : bv);

  const int m0 = blockIdx.x * 128;
  const int n0 = blockIdx.y * 128;

  const int tid  = threadIdx.x;
  const int wave = tid >> 5;
  const int lane = tid & 31;
  const int lm   = lane & 15;
  const int lh   = lane >> 4;
  const int wm   = (wave & 3) * 32;
  const int wn   = (wave >> 2) * 64;

  v8f acc[2][4] = {};

  for (int k0 = 0; k0 < E_; k0 += 32) {
    {  // stage A tile 128x32 (f32 -> f16)
      const int r  = tid >> 1;
      const int qb = (tid & 1) * 4;
      const f4* src = (const f4*)(X + (size_t)(m0 + r) * E_ + k0);
#pragma unroll
      for (int i = 0; i < 4; ++i) {
        f4 v = src[qb + i];
        _Float16* dst = &As[r * 32 + (qb + i) * 4];
        dst[0] = (_Float16)v[0]; dst[1] = (_Float16)v[1];
        dst[2] = (_Float16)v[2]; dst[3] = (_Float16)v[3];
      }
    }
    {  // stage B tile 128x32 (rows of W, since out = X @ W^T)
      const int r  = tid >> 1;
      const int qb = (tid & 1) * 4;
      const f4* src = (const f4*)(W + (size_t)(n0 + r) * E_ + k0);
#pragma unroll
      for (int i = 0; i < 4; ++i) {
        f4 v = src[qb + i];
        _Float16* dst = &Bs[r * 32 + (qb + i) * 4];
        dst[0] = (_Float16)v[0]; dst[1] = (_Float16)v[1];
        dst[2] = (_Float16)v[2]; dst[3] = (_Float16)v[3];
      }
    }
    __syncthreads();

    v16h aF[2], bF[4];
#pragma unroll
    for (int i = 0; i < 2; ++i) {  // A frag: lane=row, 16 contiguous K at h*16
      const int row = wm + i * 16 + lm;
      aF[i] = *(const v16h*)&As[row * 32 + lh * 16];
    }
#pragma unroll
    for (int j = 0; j < 4; ++j) {  // B frag: lane=col, 16 contiguous K at h*16
      const int col = wn + j * 16 + lm;
      bF[j] = *(const v16h*)&Bs[col * 32 + lh * 16];
    }
#pragma unroll
    for (int i = 0; i < 2; ++i)
#pragma unroll
      for (int j = 0; j < 4; ++j)
        acc[i][j] = wmma_f16(aF[i], bF[j], acc[i][j]);
    __syncthreads();
  }

  // epilogue: bias (+ q scaling), scatter to head layouts
#pragma unroll
  for (int j = 0; j < 4; ++j) {
    const int n = n0 + wn + j * 16 + lm;  // output feature (h*64+d)
    const float bias = bb[n];
    const int h = n >> 6;
    const int d = n & 63;
#pragma unroll
    for (int i = 0; i < 2; ++i) {
      const int mbase = m0 + wm + i * 16 + lh * 8;  // 8 consecutive rows
      const int bidx  = mbase >> 10;
      const int tbase = mbase & 1023;
      if (which == 2) {  // V transposed: [B,H,D,T] -> one packed 16B store
        v8h pk;
#pragma unroll
        for (int r = 0; r < 8; ++r) pk[r] = (_Float16)(acc[i][j][r] + bias);
        *(v8h*)&Vt[(size_t)((bidx * H_ + h) * D_ + d) * T_ + tbase] = pk;
      } else {
        _Float16* __restrict__ dst = (which == 0) ? Qo : Ko;
        const float sc = (which == 0) ? 0.125f : 1.0f;  // D^-0.5 = 1/8
#pragma unroll
        for (int r = 0; r < 8; ++r)
          dst[(size_t)((bidx * H_ + h) * T_ + tbase + r) * D_ + d] =
              (_Float16)((acc[i][j][r] + bias) * sc);
      }
    }
  }
}

// ---------------------------------------------------------------------------
// Kernel 2: flash attention with TDM double-buffered K/V tiles in LDS.
// Per wave: 16 query rows, online softmax over 32-key blocks.
// Each wave DMAs its slice of the K tile (contiguous) and V^T tile (2D
// strided) for the NEXT block while computing the current one.
// grid = (T/128, B*H), block = 256 (8 waves).
// ---------------------------------------------------------------------------
__global__ __launch_bounds__(256) void attn_kernel(
    const _Float16* __restrict__ Q, const _Float16* __restrict__ K,
    const _Float16* __restrict__ Vt, const float* __restrict__ mask,
    _Float16* __restrict__ ctx) {
  __shared__ _Float16 Ks[2][32 * 64];  // [buf][key][d]
  __shared__ _Float16 Vs[2][64 * 32];  // [buf][d][key]
  __shared__ _Float16 Ps[8][16 * 32];  // per-wave P staging

  const int bh   = blockIdx.y;
  const int b    = bh >> 4;
  const int h    = bh & 15;
  const int tid  = threadIdx.x;
  const int wave = tid >> 5;
  const int lane = tid & 31;
  const int lm   = lane & 15;
  const int lh   = lane >> 4;
  const int q0   = blockIdx.x * 128 + wave * 16;
  const int wv   = __builtin_amdgcn_readfirstlane(wave);  // uniform for TDM D#

  const _Float16* __restrict__ Qbh = Q + (size_t)bh * T_ * D_;
  const _Float16* __restrict__ Kbh = K + (size_t)bh * T_ * D_;
  const _Float16* __restrict__ Vbh = Vt + (size_t)bh * D_ * T_;
  const float* __restrict__ Mb = mask + (size_t)b * T_ * T_;

  v16h qa[2];  // Q rows as A frags, d-halves 0..31 / 32..63
#pragma unroll
  for (int dh = 0; dh < 2; ++dh) {
    const _Float16* p = Qbh + (size_t)(q0 + lm) * D_ + dh * 32;
    qa[dh] = *(const v16h*)(p + lh * 16);
  }

  v8f o[4] = {};
  float mrun[8], lrun[8];
#pragma unroll
  for (int r = 0; r < 8; ++r) { mrun[r] = -__builtin_inff(); lrun[r] = 0.0f; }

  _Float16* pw = &Ps[wave][0];

  // Issue this wave's TDM slices for key block kb into buffer bf:
  //  - K slice: 4 key rows x 64 d, fully contiguous (1 row of 256 elems)
  //  - V slice: 8 d rows x 32 keys, row stride T (true 2D tile)
  auto issue_tiles = [&](int kb, int bf) {
    tdm_load_2d_f16(lds_off(&Ks[bf][wv * 4 * 64]),
                    (unsigned long long)(Kbh + (size_t)(kb + wv * 4) * D_),
                    /*tile*/ 256, 1,
                    /*tensor*/ (unsigned)(T_ * D_), 1,
                    /*stride0*/ (unsigned long long)(T_ * D_));
    tdm_load_2d_f16(lds_off(&Vs[bf][wv * 8 * 32]),
                    (unsigned long long)(Vbh + (size_t)(wv * 8) * T_ + kb),
                    /*tile*/ 32, 8,
                    /*tensor*/ (unsigned)T_, (unsigned)D_,
                    /*stride0*/ (unsigned long long)T_);
  };

  int buf = 0;
  issue_tiles(0, 0);

  for (int kb = 0; kb < T_; kb += 32) {
    __builtin_amdgcn_s_wait_tensorcnt(0);  // my slices for `buf` landed
    __syncthreads();                       // everyone's slices landed
    if (kb + 32 < T_) {
      issue_tiles(kb + 32, buf ^ 1);       // overlap next DMA with compute
      __builtin_prefetch(Mb + (size_t)(q0 + lm) * T_ + kb + 32, 0, 0);
    }

    // S = Q K^T  (two 16-key subtiles, d=64 in two K=32 WMMA steps)
    v8f s[2];
#pragma unroll
    for (int nt = 0; nt < 2; ++nt) {
      v8f st = {};
#pragma unroll
      for (int dh = 0; dh < 2; ++dh) {
        const v16h kf =
            *(const v16h*)&Ks[buf][(nt * 16 + lm) * 64 + dh * 32 + lh * 16];
        st = wmma_f16(qa[dh], kf, st);
      }
      s[nt] = st;
    }
    // + attention mask  (C layout: lane holds col lm, rows r + 8*lh)
#pragma unroll
    for (int nt = 0; nt < 2; ++nt)
#pragma unroll
      for (int r = 0; r < 8; ++r)
        s[nt][r] += Mb[(size_t)(q0 + r + 8 * lh) * T_ + kb + nt * 16 + lm];

    // online softmax: row reductions across the 16-lane row group
    float scale[8];
#pragma unroll
    for (int r = 0; r < 8; ++r) {
      float t = fmaxf(s[0][r], s[1][r]);
      t = fmaxf(t, __shfl_xor(t, 1, 32));
      t = fmaxf(t, __shfl_xor(t, 2, 32));
      t = fmaxf(t, __shfl_xor(t, 4, 32));
      t = fmaxf(t, __shfl_xor(t, 8, 32));
      const float mnew = fmaxf(mrun[r], t);
      scale[r] = __expf(mrun[r] - mnew);
      s[0][r] = __expf(s[0][r] - mnew);
      s[1][r] = __expf(s[1][r] - mnew);
      float ps = s[0][r] + s[1][r];
      ps += __shfl_xor(ps, 1, 32);
      ps += __shfl_xor(ps, 2, 32);
      ps += __shfl_xor(ps, 4, 32);
      ps += __shfl_xor(ps, 8, 32);
      lrun[r] = lrun[r] * scale[r] + ps;
      mrun[r] = mnew;
    }
#pragma unroll
    for (int dt = 0; dt < 4; ++dt)
#pragma unroll
      for (int r = 0; r < 8; ++r) o[dt][r] *= scale[r];

    // P (f32 C-layout) -> LDS f16 -> reload as A frag (same-wave DS in-order)
#pragma unroll
    for (int nt = 0; nt < 2; ++nt)
#pragma unroll
      for (int r = 0; r < 8; ++r)
        pw[(r + 8 * lh) * 32 + nt * 16 + lm] = (_Float16)s[nt][r];

    const v16h pa = *(const v16h*)&pw[lm * 32 + lh * 16];

    // O += P V   (V^T tile: lane=d column, 16 contiguous keys at h*16)
#pragma unroll
    for (int dt = 0; dt < 4; ++dt) {
      const v16h vf = *(const v16h*)&Vs[buf][(dt * 16 + lm) * 32 + lh * 16];
      o[dt] = wmma_f16(pa, vf, o[dt]);
    }
    buf ^= 1;
  }

  // normalize and store ctx as [B,T,H,D] f16 (row-major [4096,1024])
#pragma unroll
  for (int dt = 0; dt < 4; ++dt) {
    const int d = dt * 16 + lm;
#pragma unroll
    for (int r = 0; r < 8; ++r) {
      const int t = q0 + r + 8 * lh;
      ctx[(size_t)((b * T_ + t) * H_ + h) * D_ + d] =
          (_Float16)(o[dt][r] / lrun[r]);
    }
  }
}

// ---------------------------------------------------------------------------
// Kernel 3: output projection.  out = ctx @ Wo^T + bo, f32 result.
// Wave tile 32x64 (2x4 WMMA); direct global fragment loads (Wo L2-resident).
// ---------------------------------------------------------------------------
__global__ __launch_bounds__(256) void oproj_kernel(
    const _Float16* __restrict__ Cx, const float* __restrict__ Wo,
    const float* __restrict__ bo, float* __restrict__ out) {
  const int tid  = threadIdx.x;
  const int wave = tid >> 5;
  const int lane = tid & 31;
  const int lm   = lane & 15;
  const int lh   = lane >> 4;
  const int m0   = blockIdx.x * 128 + (wave & 3) * 32;
  const int n0   = blockIdx.y * 128 + (wave >> 2) * 64;

  v8f acc[2][4] = {};
  for (int k0 = 0; k0 < E_; k0 += 32) {
    v16h aF[2], bF[4];
#pragma unroll
    for (int i = 0; i < 2; ++i) {
      const _Float16* p = Cx + (size_t)(m0 + i * 16 + lm) * E_ + k0;
      aF[i] = *(const v16h*)(p + lh * 16);
    }
#pragma unroll
    for (int j = 0; j < 4; ++j) {
      const f4* wp =
          (const f4*)(Wo + (size_t)(n0 + j * 16 + lm) * E_ + k0 + lh * 16);
      v16h bv;
#pragma unroll
      for (int q = 0; q < 4; ++q) {
        f4 v = wp[q];
        bv[q * 4 + 0] = (_Float16)v[0]; bv[q * 4 + 1] = (_Float16)v[1];
        bv[q * 4 + 2] = (_Float16)v[2]; bv[q * 4 + 3] = (_Float16)v[3];
      }
      bF[j] = bv;
    }
#pragma unroll
    for (int i = 0; i < 2; ++i)
#pragma unroll
      for (int j = 0; j < 4; ++j)
        acc[i][j] = wmma_f16(aF[i], bF[j], acc[i][j]);
  }
#pragma unroll
  for (int j = 0; j < 4; ++j) {
    const int n = n0 + j * 16 + lm;
    const float bias = bo[n];
#pragma unroll
    for (int i = 0; i < 2; ++i) {
      const int mb = m0 + i * 16 + lh * 8;
#pragma unroll
      for (int r = 0; r < 8; ++r)
        out[(size_t)(mb + r) * E_ + n] = acc[i][j][r] + bias;
    }
  }
}

// ---------------------------------------------------------------------------
extern "C" void kernel_launch(void* const* d_in, const int* in_sizes, int n_in,
                              void* d_out, int out_size, void* d_ws,
                              size_t ws_size, hipStream_t stream) {
  (void)in_sizes; (void)n_in; (void)out_size; (void)ws_size;

  const float* X    = (const float*)d_in[0];
  const float* mask = (const float*)d_in[1];
  const float* Wq   = (const float*)d_in[2];
  const float* bq   = (const float*)d_in[3];
  const float* Wk   = (const float*)d_in[4];
  const float* bk   = (const float*)d_in[5];
  const float* Wv   = (const float*)d_in[6];
  const float* bv   = (const float*)d_in[7];
  const float* Wo   = (const float*)d_in[8];
  const float* bo   = (const float*)d_in[9];
  float* out = (float*)d_out;

  char* w = (char*)d_ws;
  _Float16* Qf = (_Float16*)(w);                        // 8 MB  [B,H,T,D]
  _Float16* Kf = (_Float16*)(w + (size_t)8  * 1048576); // 8 MB  [B,H,T,D]
  _Float16* Vt = (_Float16*)(w + (size_t)16 * 1048576); // 8 MB  [B,H,D,T]
  _Float16* Cx = (_Float16*)(w + (size_t)24 * 1048576); // 8 MB  [B,T,H,D]

  qkv_proj_kernel<<<dim3(32, 8, 3), 256, 0, stream>>>(X, Wq, bq, Wk, bk, Wv,
                                                      bv, Qf, Kf, Vt);
  attn_kernel<<<dim3(8, 64), 256, 0, stream>>>(Qf, Kf, Vt, mask, Cx);
  oproj_kernel<<<dim3(32, 8), 256, 0, stream>>>(Cx, Wo, bo, out);
}
